// BigramLanguageModel_47845935677978
// MI455X (gfx1250) — matrix operand order repaired and checked by reading, not compiled, not run on hardware
//
#include <hip/hip_runtime.h>

// ---------------------------------------------------------------------------
// GPT forward on MI455X (gfx1250): bf16 WMMA GEMMs with async-to-LDS double
// buffering + fp32 LN/softmax/loss.
// ---------------------------------------------------------------------------

typedef __attribute__((ext_vector_type(16))) __bf16 bf16x16;
typedef __attribute__((ext_vector_type(8)))  float  f32x8;

// ---- constants for this model ----
constexpr int  Bc  = 4;
constexpr int  Tc  = 1024;
constexpr int  Dc  = 1024;
constexpr int  Hc  = 16;
constexpr int  HSc = 64;
constexpr int  Lc  = 6;
constexpr int  FFc = 4096;
constexpr int  Vc  = 32000;
constexpr int  BTc = Bc * Tc;

// GEMM tiling
constexpr int BM = 128, BN = 128, BK = 32;
constexpr int SAP = BK + 8;    // LDS pitch for A tile (ushorts) -> 80B rows (16B aligned)
constexpr int SBP = BN + 8;    // LDS pitch for B tile (ushorts) -> 272B rows (16B aligned)

__device__ __forceinline__ unsigned short f2bf(float f) {
  union { float f; unsigned u; } cv; cv.f = f;
  unsigned r = (cv.u + 0x7FFFu + ((cv.u >> 16) & 1u)) >> 16;
  return (unsigned short)r;
}

// gfx1250 async copy: memory -> LDS, tracked by ASYNCcnt (no VGPR round-trip)
__device__ __forceinline__ void async_ld_b128(unsigned lds_byte_addr, const void* gaddr) {
  asm volatile("global_load_async_to_lds_b128 %0, %1, off"
               :: "v"(lds_byte_addr), "v"(gaddr)
               : "memory");
}
template <int N>
__device__ __forceinline__ void wait_asynccnt_le() {
  asm volatile("s_wait_asynccnt %0" :: "n"(N) : "memory");
}

__device__ __forceinline__ float block_reduce_sum(float v, float* red) {
  int t = threadIdx.x;
  red[t] = v; __syncthreads();
  for (int s = 128; s > 0; s >>= 1) { if (t < s) red[t] += red[t + s]; __syncthreads(); }
  float r = red[0]; __syncthreads();
  return r;
}
__device__ __forceinline__ float block_reduce_max(float v, float* red) {
  int t = threadIdx.x;
  red[t] = v; __syncthreads();
  for (int s = 128; s > 0; s >>= 1) { if (t < s) red[t] = fmaxf(red[t], red[t + s]); __syncthreads(); }
  float r = red[0]; __syncthreads();
  return r;
}

// ---------------------------------------------------------------------------
// Generic batched bf16 GEMM:  C[z] = epilogue( scale * A[z] x B[z] )
//   A: M x K bf16 row-major (lda), batched via (sAb,sAh), z = b*Hdiv + h
//   B: K x N bf16 row-major (ldb) or, if TRANSB, stored N x K (B[k][n]=Bs[n*ldb+k])
//   C: M x N fp32 (or bf16 if OUTBF16), ldc
//   epilogue: v = acc*scale + bias[n] (+ residual[m*ldc+n]) (+ ReLU)
// Requires: M % 128 == 0, K % 32 == 0, N % 16 == 0, 16B-aligned bf16 bases.
// Async-to-LDS double-buffered pipeline; B-tile columns >= N are loaded
// (in-bounds workspace garbage) but masked in the epilogue.
// ---------------------------------------------------------------------------
template <bool TRANSB, bool RELU, bool OUTBF16>
__global__ __launch_bounds__(256)
void gemm_wmma_bf16(int M, int N, int K,
                    const unsigned short* __restrict__ A, int lda, long long sAb, long long sAh,
                    const unsigned short* __restrict__ B, int ldb, long long sBb, long long sBh,
                    void* __restrict__ C, int ldc, long long sCb, long long sCh,
                    const float* __restrict__ bias, const float* __restrict__ residual,
                    float scale, int Hdiv) {
  __shared__ __attribute__((aligned(16))) unsigned short sA[2][BM * SAP];
  __shared__ __attribute__((aligned(16))) unsigned short sB[2][BK * SBP];

  const int z  = blockIdx.z;
  const int zb = z / Hdiv;
  const int zh = z - zb * Hdiv;
  A += (size_t)(zb * sAb + zh * sAh);
  B += (size_t)(zb * sBb + zh * sBh);
  unsigned short* Cb16 = (unsigned short*)C + (size_t)(zb * sCb + zh * sCh);
  float*          Cf32 = (float*)C          + (size_t)(zb * sCb + zh * sCh);

  const int tid  = threadIdx.x;
  const int lane = tid & 31;
  const int wave = tid >> 5;
  const int wm   = (wave >> 2) * 64;   // wave M offset (2 waves along M)
  const int wn   = (wave & 3) * 32;    // wave N offset (4 waves along N)
  const int m0   = blockIdx.y * BM;
  const int n0   = blockIdx.x * BN;

  f32x8 acc[4][2];
  #pragma unroll
  for (int mi = 0; mi < 4; ++mi)
    #pragma unroll
    for (int ni = 0; ni < 2; ++ni) acc[mi][ni] = (f32x8){0.f,0.f,0.f,0.f,0.f,0.f,0.f,0.f};

  // per-thread staging coordinates
  const int arow = tid >> 1, acol = (tid & 1) << 4;   // A: 128 rows x 2 chunks of 16 bf16
  const int brow = tid >> 3, bcol = (tid & 7) << 4;   // B normal: 32 rows x 8 chunks
  const int tcol = tid >> 1, tkch = (tid & 1) << 4;   // B trans: 128 cols x 2 k-chunks

  constexpr int INFLIGHT = TRANSB ? 2 : 4;            // async instrs per tile per wave

  auto issue_tile = [&](int k0, int buf) {
    { // A tile: 32 contiguous bytes per thread -> 2x b128 async
      const unsigned short* g = A + (size_t)(m0 + arow) * lda + k0 + acol;
      unsigned l = (unsigned)(uintptr_t)&sA[buf][arow * SAP + acol];
      async_ld_b128(l,      g);
      async_ld_b128(l + 16, g + 8);
    }
    if (!TRANSB) { // B tile: 32 contiguous bytes per thread -> 2x b128 async
      const unsigned short* g = B + (size_t)(k0 + brow) * ldb + n0 + bcol;
      unsigned l = (unsigned)(uintptr_t)&sB[buf][brow * SBP + bcol];
      async_ld_b128(l,      g);
      async_ld_b128(l + 16, g + 8);
    }
  };

  auto stage_b_trans = [&](int k0, int buf) {  // VGPR transpose path (scores GEMM)
    union { uint4 q[2]; unsigned short s[16]; } tmp;
    tmp.q[0] = make_uint4(0, 0, 0, 0); tmp.q[1] = make_uint4(0, 0, 0, 0);
    if (n0 + tcol < N) {
      const uint4* gp = reinterpret_cast<const uint4*>(B + (size_t)(n0 + tcol) * ldb + k0 + tkch);
      tmp.q[0] = gp[0]; tmp.q[1] = gp[1];
    }
    #pragma unroll
    for (int j = 0; j < 16; ++j) sB[buf][(tkch + j) * SBP + tcol] = tmp.s[j];
  };

  const int nK = K >> 5;
  issue_tile(0, 0);

  for (int kb = 0; kb < nK; ++kb) {
    const int buf = kb & 1;

    if (TRANSB) stage_b_trans(kb << 5, buf);

    if (kb + 1 < nK) {
      issue_tile((kb + 1) << 5, buf ^ 1);
      if (kb + 2 < nK)
        __builtin_prefetch(A + (size_t)(m0 + arow) * lda + ((kb + 2) << 5) + acol, 0, 1);
      wait_asynccnt_le<INFLIGHT>();   // tile kb's async copies complete
    } else {
      wait_asynccnt_le<0>();
    }
    __syncthreads();                  // all waves' tile-kb data visible in LDS

    // ---- fragments from LDS per documented wave32 layouts ----
    union Frag { unsigned u[8]; bf16x16 v; };
    Frag af[4], bfm[2];
    {
      const int m    = lane & 15;
      const int half = lane >> 4;
      #pragma unroll
      for (int mi = 0; mi < 4; ++mi) {
        const unsigned short* rp = &sA[buf][(wm + mi * 16 + m) * SAP];
        #pragma unroll
        for (int v = 0; v < 8; ++v) {
          int kk = ((v < 4) ? (2 * v) : (16 + 2 * (v - 4))) + 8 * half;
          af[mi].u[v] = *reinterpret_cast<const unsigned*>(rp + kk);
        }
      }
      #pragma unroll
      for (int ni = 0; ni < 2; ++ni) {
        const unsigned short* rp = &sB[buf][wn + ni * 16 + m];
        #pragma unroll
        for (int v = 0; v < 8; ++v) {
          int kk = ((v < 4) ? (2 * v) : (16 + 2 * (v - 4))) + 8 * half;
          bfm[ni].u[v] = (unsigned)rp[kk * SBP] | ((unsigned)rp[(kk + 1) * SBP] << 16);
        }
      }
    }

    #pragma unroll
    for (int mi = 0; mi < 4; ++mi)
      #pragma unroll
      for (int ni = 0; ni < 2; ++ni)
        acc[mi][ni] = __builtin_amdgcn_wmma_f32_16x16x32_bf16(
            false, af[mi].v, false, bfm[ni].v, (short)0, acc[mi][ni], false, false);

    __syncthreads();                  // done reading buf before it is overwritten
  }

  // ---- epilogue: C/D layout lanes 0-15 -> M=r, lanes 16-31 -> M=r+8 ----
  const int half = lane >> 4;
  #pragma unroll
  for (int mi = 0; mi < 4; ++mi) {
    #pragma unroll
    for (int ni = 0; ni < 2; ++ni) {
      const int gcol = n0 + wn + ni * 16 + (lane & 15);
      if (gcol < N) {
        const float bv = bias ? bias[gcol] : 0.0f;
        #pragma unroll
        for (int r = 0; r < 8; ++r) {
          const int grow = m0 + wm + mi * 16 + half * 8 + r;
          const size_t ci = (size_t)grow * ldc + gcol;
          float v = acc[mi][ni][r] * scale + bv;
          if (residual) v += residual[ci];
          if (RELU) v = fmaxf(v, 0.0f);
          if (OUTBF16) Cb16[ci] = f2bf(v);
          else         Cf32[ci] = v;
        }
      }
    }
  }
}

// ---------------------------------------------------------------------------
// Elementwise / reduction kernels
// ---------------------------------------------------------------------------
__global__ void cvt_bf16_k(const float* __restrict__ s, unsigned short* __restrict__ d, size_t n) {
  size_t i = (size_t)blockIdx.x * 256 + threadIdx.x;
  if (i < n) d[i] = f2bf(s[i]);
}

__global__ void embed_k(const int* __restrict__ idx, const float* __restrict__ tok,
                        const float* __restrict__ pos, float* __restrict__ x,
                        int T, int D, size_t n) {
  size_t i = (size_t)blockIdx.x * 256 + threadIdx.x;
  if (i >= n) return;
  int d = (int)(i % D);
  size_t bt = i / D;
  int t = (int)(bt % T);
  x[i] = tok[(size_t)idx[bt] * D + d] + pos[(size_t)t * D + d];
}

// one row (length 1024) per 256-thread block, fp32 in -> bf16 out
__global__ void layernorm_bf16_k(const float* __restrict__ x, const float* __restrict__ g,
                                 const float* __restrict__ b, unsigned short* __restrict__ out,
                                 int D) {
  __shared__ float red[256];
  const int row = blockIdx.x, tid = threadIdx.x;
  const float* xr = x + (size_t)row * D;
  float v[4], s = 0.f;
  #pragma unroll
  for (int j = 0; j < 4; ++j) { v[j] = xr[tid + j * 256]; s += v[j]; }
  const float mean = block_reduce_sum(s, red) * (1.0f / 1024.0f);
  float sq = 0.f;
  #pragma unroll
  for (int j = 0; j < 4; ++j) { float d = v[j] - mean; sq += d * d; }
  const float var = block_reduce_sum(sq, red) * (1.0f / 1024.0f);
  const float rs = rsqrtf(var + 1e-5f);
  #pragma unroll
  for (int j = 0; j < 4; ++j) {
    const int c = tid + j * 256;
    out[(size_t)row * D + c] = f2bf((v[j] - mean) * rs * g[c] + b[c]);
  }
}

// causal softmax over one score row; rewrites the row in place as bf16 probs
// (probs row (bh,t) lives at ushort index ((bh*T+t)*2*T + u) of the same buffer)
__global__ void softmax_causal_k(float* __restrict__ S, int T) {
  __shared__ float red[256];
  const int row = blockIdx.x;           // bh*T + t
  const int t = row % T;
  const int tid = threadIdx.x;
  float* sr = S + (size_t)row * T;
  float v[4];
  float mx = -3.0e38f;
  #pragma unroll
  for (int j = 0; j < 4; ++j) {
    const int c = tid + j * 256;
    v[j] = sr[c];
    if (c <= t) mx = fmaxf(mx, v[j]);
  }
  mx = block_reduce_max(mx, red);
  float e[4], s = 0.f;
  #pragma unroll
  for (int j = 0; j < 4; ++j) {
    const int c = tid + j * 256;
    e[j] = (c <= t) ? __expf(v[j] - mx) : 0.0f;
    s += e[j];
  }
  s = block_reduce_sum(s, red);
  const float inv = 1.0f / s;
  unsigned short* pr = (unsigned short*)sr;   // stays inside this row's fp32 span
  #pragma unroll
  for (int j = 0; j < 4; ++j) pr[tid + j * 256] = f2bf(e[j] * inv);
}

__global__ void loss_rows_k(const float* __restrict__ logits, const int* __restrict__ tgt,
                            float* __restrict__ rl, int V) {
  __shared__ float red[256];
  const int row = blockIdx.x, tid = threadIdx.x;
  const float* lr = logits + (size_t)row * V;
  float mx = -3.0e38f;
  for (int c = tid; c < V; c += 256) mx = fmaxf(mx, lr[c]);
  mx = block_reduce_max(mx, red);
  float s = 0.f;
  for (int c = tid; c < V; c += 256) s += __expf(lr[c] - mx);
  s = block_reduce_sum(s, red);
  if (tid == 0) rl[row] = -(lr[tgt[row]] - mx - __logf(s));
}

__global__ void loss_reduce_k(const float* __restrict__ rl, float* __restrict__ out, int n) {
  __shared__ float red[256];
  const int tid = threadIdx.x;
  float s = 0.f;
  for (int i = tid; i < n; i += 256) s += rl[i];
  s = block_reduce_sum(s, red);
  if (tid == 0) out[0] = s / (float)n;
}

// ---------------------------------------------------------------------------
// Host side
// ---------------------------------------------------------------------------
static inline unsigned cdiv(size_t a, size_t b) { return (unsigned)((a + b - 1) / b); }

extern "C" void kernel_launch(void* const* d_in, const int* in_sizes, int n_in,
                              void* d_out, int out_size, void* d_ws, size_t ws_size,
                              hipStream_t stream) {
  (void)in_sizes; (void)n_in; (void)out_size; (void)ws_size;

  const int*   idx     = (const int*)  d_in[0];
  const int*   targets = (const int*)  d_in[1];
  const float* tok_emb = (const float*)d_in[2];
  const float* pos_emb = (const float*)d_in[3];
  const float* Wq      = (const float*)d_in[4];
  const float* Wk      = (const float*)d_in[5];
  const float* Wv      = (const float*)d_in[6];
  const float* proj_w  = (const float*)d_in[7];
  const float* proj_b  = (const float*)d_in[8];
  const float* ln1_g   = (const float*)d_in[9];
  const float* ln1_b   = (const float*)d_in[10];
  const float* ln2_g   = (const float*)d_in[11];
  const float* ln2_b   = (const float*)d_in[12];
  const float* ff_w1   = (const float*)d_in[13];
  const float* ff_b1   = (const float*)d_in[14];
  const float* ff_w2   = (const float*)d_in[15];
  const float* ff_b2   = (const float*)d_in[16];
  const float* lnf_g   = (const float*)d_in[17];
  const float* lnf_b   = (const float*)d_in[18];
  const float* head_w  = (const float*)d_in[19];
  const float* head_b  = (const float*)d_in[20];

  float* logits = (float*)d_out;

  // ---- carve workspace ----
  char* w = (char*)d_ws;
  auto aus = [&](size_t n) { unsigned short* p = (unsigned short*)w; w += n * sizeof(unsigned short); return p; };
  auto af32 = [&](size_t n) { float* p = (float*)w; w += n * sizeof(float); return p; };

  const size_t nQKVw = (size_t)Lc * Dc * Dc;
  const size_t nFF1w = (size_t)Lc * Dc * FFc;
  const size_t nFF2w = (size_t)Lc * FFc * Dc;
  const size_t nHdw  = (size_t)Dc * Vc;
  const size_t nAct  = (size_t)BTc * Dc;
  const size_t nFFa  = (size_t)BTc * FFc;
  const size_t nScr  = (size_t)Bc * Hc * Tc * Tc;

  unsigned short* wqb = aus(nQKVw);
  unsigned short* wkb = aus(nQKVw);
  unsigned short* wvb = aus(nQKVw);
  unsigned short* wpb = aus(nQKVw);
  unsigned short* w1b = aus(nFF1w);
  unsigned short* w2b = aus(nFF2w);
  unsigned short* whb = aus(nHdw);
  unsigned short* hbuf = aus(nAct);
  unsigned short* qbuf = aus(nAct);
  unsigned short* kbuf = aus(nAct);
  unsigned short* vbuf = aus(nAct);
  unsigned short* obuf = aus(nAct);
  unsigned short* ffb  = aus(nFFa);
  float* x      = af32(nAct);
  float* scores = af32(nScr);      // rewritten in place as bf16 probs by softmax
  float* rloss  = af32(BTc);

  const dim3 blk(256);

  // ---- weight conversion fp32 -> bf16 ----
  cvt_bf16_k<<<cdiv(nQKVw, 256), blk, 0, stream>>>(Wq, wqb, nQKVw);
  cvt_bf16_k<<<cdiv(nQKVw, 256), blk, 0, stream>>>(Wk, wkb, nQKVw);
  cvt_bf16_k<<<cdiv(nQKVw, 256), blk, 0, stream>>>(Wv, wvb, nQKVw);
  cvt_bf16_k<<<cdiv(nQKVw, 256), blk, 0, stream>>>(proj_w, wpb, nQKVw);
  cvt_bf16_k<<<cdiv(nFF1w, 256), blk, 0, stream>>>(ff_w1, w1b, nFF1w);
  cvt_bf16_k<<<cdiv(nFF2w, 256), blk, 0, stream>>>(ff_w2, w2b, nFF2w);
  cvt_bf16_k<<<cdiv(nHdw, 256), blk, 0, stream>>>(head_w, whb, nHdw);

  // ---- embeddings ----
  embed_k<<<cdiv(nAct, 256), blk, 0, stream>>>(idx, tok_emb, pos_emb, x, Tc, Dc, nAct);

  const float attn_scale = 0.03125f;   // D^-0.5 = 1/32 (reference scales by full D)
  unsigned short* probs = (unsigned short*)scores;  // bf16 view, lda = 2*T

  for (int l = 0; l < Lc; ++l) {
    // LN1 -> bf16
    layernorm_bf16_k<<<BTc, blk, 0, stream>>>(x, ln1_g + (size_t)l * Dc, ln1_b + (size_t)l * Dc, hbuf, Dc);

    // Q, K, V : [BT,D] x [D,D] -> bf16
    {
      dim3 g(Dc / BN, BTc / BM, 1);
      gemm_wmma_bf16<false, false, true><<<g, blk, 0, stream>>>(
          BTc, Dc, Dc, hbuf, Dc, 0, 0, wqb + (size_t)l * Dc * Dc, Dc, 0, 0,
          qbuf, Dc, 0, 0, nullptr, nullptr, 1.0f, 1);
      gemm_wmma_bf16<false, false, true><<<g, blk, 0, stream>>>(
          BTc, Dc, Dc, hbuf, Dc, 0, 0, wkb + (size_t)l * Dc * Dc, Dc, 0, 0,
          kbuf, Dc, 0, 0, nullptr, nullptr, 1.0f, 1);
      gemm_wmma_bf16<false, false, true><<<g, blk, 0, stream>>>(
          BTc, Dc, Dc, hbuf, Dc, 0, 0, wvb + (size_t)l * Dc * Dc, Dc, 0, 0,
          vbuf, Dc, 0, 0, nullptr, nullptr, 1.0f, 1);
    }

    // scores[bh] = scale * q[bh] x k[bh]^T : [T,HS] x [HS,T]
    {
      dim3 g(Tc / BN, Tc / BM, Bc * Hc);
      gemm_wmma_bf16<true, false, false><<<g, blk, 0, stream>>>(
          Tc, Tc, HSc,
          qbuf, Dc, (long long)Tc * Dc, (long long)HSc,
          kbuf, Dc, (long long)Tc * Dc, (long long)HSc,
          scores, Tc, (long long)Hc * Tc * Tc, (long long)Tc * Tc,
          nullptr, nullptr, attn_scale, Hc);
    }

    // causal softmax (in place, fp32 -> bf16 probs)
    softmax_causal_k<<<Bc * Hc * Tc, blk, 0, stream>>>(scores, Tc);

    // o[bh] = P[bh] x v[bh] : [T,T] x [T,HS] -> bf16 (written interleaved per head)
    {
      dim3 g(1, Tc / BM, Bc * Hc);
      gemm_wmma_bf16<false, false, true><<<g, blk, 0, stream>>>(
          Tc, HSc, Tc,
          probs, 2 * Tc, (long long)Hc * 2 * Tc * Tc, (long long)2 * Tc * Tc,
          vbuf, Dc, (long long)Tc * Dc, (long long)HSc,
          obuf, Dc, (long long)Tc * Dc, (long long)HSc,
          nullptr, nullptr, 1.0f, Hc);
    }

    // x = x + o x Wproj + proj_b
    {
      dim3 g(Dc / BN, BTc / BM, 1);
      gemm_wmma_bf16<false, false, false><<<g, blk, 0, stream>>>(
          BTc, Dc, Dc, obuf, Dc, 0, 0, wpb + (size_t)l * Dc * Dc, Dc, 0, 0,
          x, Dc, 0, 0, proj_b + (size_t)l * Dc, x, 1.0f, 1);
    }

    // LN2 -> bf16
    layernorm_bf16_k<<<BTc, blk, 0, stream>>>(x, ln2_g + (size_t)l * Dc, ln2_b + (size_t)l * Dc, hbuf, Dc);

    // ff = relu(h x W1 + b1) -> bf16
    {
      dim3 g(FFc / BN, BTc / BM, 1);
      gemm_wmma_bf16<false, true, true><<<g, blk, 0, stream>>>(
          BTc, FFc, Dc, hbuf, Dc, 0, 0, w1b + (size_t)l * Dc * FFc, FFc, 0, 0,
          ffb, FFc, 0, 0, ff_b1 + (size_t)l * FFc, nullptr, 1.0f, 1);
    }

    // x = x + ff x W2 + b2
    {
      dim3 g(Dc / BN, BTc / BM, 1);
      gemm_wmma_bf16<false, false, false><<<g, blk, 0, stream>>>(
          BTc, Dc, FFc, ffb, FFc, 0, 0, w2b + (size_t)l * FFc * Dc, Dc, 0, 0,
          x, Dc, 0, 0, ff_b2 + (size_t)l * Dc, x, 1.0f, 1);
    }
  }

  // final LN -> bf16
  layernorm_bf16_k<<<BTc, blk, 0, stream>>>(x, lnf_g, lnf_b, hbuf, Dc);

  // logits = h x head_w + head_b  -> d_out (fp32)
  {
    dim3 g(Vc / BN, BTc / BM, 1);
    gemm_wmma_bf16<false, false, false><<<g, blk, 0, stream>>>(
        BTc, Vc, Dc, hbuf, Dc, 0, 0, whb, Vc, 0, 0,
        logits, Vc, 0, 0, head_b, nullptr, 1.0f, 1);
  }

  // cross-entropy loss -> d_out[BT*V]
  loss_rows_k<<<BTc, blk, 0, stream>>>(logits, targets, rloss, Vc);
  loss_reduce_k<<<1, blk, 0, stream>>>(rloss, logits + (size_t)BTc * Vc, BTc);
}
